// RelativeSelfAttention_45380624450281
// MI455X (gfx1250) — matrix operand order repaired and checked by reading, not compiled, not run on hardware
//
#include <hip/hip_runtime.h>
#include <hip/hip_bf16.h>

#define S_LEN 2048
#define D_MODEL 1024
#define NHEADS 16
#define HDIM 64

typedef __bf16 bf16_t;
typedef bf16_t v16bf __attribute__((ext_vector_type(16)));
typedef float v8f __attribute__((ext_vector_type(8)));

union Frag { v16bf v; uint4 q[2]; unsigned short u[16]; };
union Acc  { v8f  v; float f[8]; };

// ---- CDNA5 async global->LDS staging (guarded for toolchain availability) ----
#if __has_builtin(__builtin_amdgcn_global_load_async_to_lds_b128) && \
    __has_builtin(__builtin_amdgcn_s_wait_asynccnt)
#define USE_ASYNC_LDS 1
#else
#define USE_ASYNC_LDS 0
#endif

typedef int ai4 __attribute__((ext_vector_type(4)));
typedef __attribute__((address_space(1))) ai4* g_i4p;   // global int4*
typedef __attribute__((address_space(3))) ai4* l_i4p;   // LDS int4*

__device__ __forceinline__ void stage16B(const unsigned short* g, unsigned short* l){
#if USE_ASYNC_LDS
  __builtin_amdgcn_global_load_async_to_lds_b128(
      (g_i4p)(const void*)g, (l_i4p)(void*)l, 0, 0);
#else
  *(uint4*)l = *(const uint4*)g;
#endif
}

__device__ __forceinline__ void stage_wait(){
#if USE_ASYNC_LDS
  __builtin_amdgcn_s_wait_asynccnt(0);
#endif
}

__device__ __forceinline__ unsigned short f2bf(float f){
  unsigned u = __float_as_uint(f);
  u += 0x7FFFu + ((u >> 16) & 1u);          // round-to-nearest-even
  return (unsigned short)(u >> 16);
}

__device__ __forceinline__ void zero_acc(Acc& a){
#pragma unroll
  for (int v = 0; v < 8; ++v) a.f[v] = 0.0f;
}

__device__ __forceinline__ void wmma_bf16(Acc& acc, const Frag& a, const Frag& b){
  acc.v = __builtin_amdgcn_wmma_f32_16x16x32_bf16(false, a.v, false, b.v,
                                                  (short)0, acc.v, false, false);
}

// A fragment (16x32, M x K): lane m = lane&15; per ISA layout lane half selects
// k chunks [8h..8h+7] and [16+8h..16+8h+7] -> two 16B contiguous loads.
__device__ __forceinline__ void load_a(Frag& f, const unsigned short* base,
                                       int row, int ld, int k0, int half){
  const unsigned short* p = base + row * ld + k0 + 8 * half;
  f.q[0] = *(const uint4*)(p);
  f.q[1] = *(const uint4*)(p + 16);
}

// B fragment (32x16, K x N): lane n = lane&15, k chunks [8h..8h+7] and
// [16+8h..16+8h+7] (same as A) from the "n-th row" of a [N][K]-laid-out source.
__device__ __forceinline__ void load_b(Frag& f, const unsigned short* nrow, int half){
  const unsigned short* p = nrow + 8 * half;
  f.q[0] = *(const uint4*)(p);
  f.q[1] = *(const uint4*)(p + 16);
}

// ---------------- conversion kernels ----------------

__global__ void cvt_bf16(const float* __restrict__ src, unsigned short* __restrict__ dst, int n){
  int i = blockIdx.x * blockDim.x + threadIdx.x;
  if (i < n) dst[i] = f2bf(src[i]);
}

__global__ void cvt_rel(const float* __restrict__ src, unsigned short* __restrict__ dst){
  // (2S-1) x 64 -> 4096 x 64 (last row zero so the 32-row band never reads junk)
  int i = blockIdx.x * blockDim.x + threadIdx.x;
  const int n_src = (2 * S_LEN - 1) * HDIM;
  if (i < 4096 * HDIM) dst[i] = (i < n_src) ? f2bf(src[i]) : (unsigned short)0;
}

__global__ void cvt_transpose(const float* __restrict__ W, unsigned short* __restrict__ Wt){
  // W[k][n] f32 -> Wt[n][k] bf16, 1024x1024
  int i = blockIdx.x * blockDim.x + threadIdx.x;
  int n = i >> 10, k = i & 1023;
  Wt[i] = f2bf(W[k * D_MODEL + n]);
}

// ---------------- 2048x1024x1024 GEMM ----------------
// A: [2048][1024] bf16 row-major.  Bt: [1024 n][1024 k] bf16 (pre-transposed).
// mode 0: store bf16 head layout [H][S][64].  mode 1: store f32 [M][N].
__global__ __launch_bounds__(256)
void gemm2048(const unsigned short* __restrict__ A, const unsigned short* __restrict__ Bt,
              const float* __restrict__ bias, void* __restrict__ out,
              float alpha, int mode){
  __shared__ unsigned short Al[128][40];   // padded stride: conflict-free b128 reads
  __shared__ unsigned short Bl[128][40];
  const int K = 1024;
  int tid  = threadIdx.x;
  int lane = tid & 31, n16 = lane & 15, half = lane >> 4;
  int w    = tid >> 5;
  int mt   = blockIdx.x & 15;
  int nt   = blockIdx.x >> 4;
  int m0   = mt * 128, n0 = nt * 128;
  int wm   = w & 3, wn = w >> 2;
  int srow = tid >> 1, sch = (tid & 1) * 16;

  Acc acc[2][4];
#pragma unroll
  for (int i = 0; i < 2; ++i)
#pragma unroll
    for (int j = 0; j < 4; ++j) zero_acc(acc[i][j]);

  for (int k0 = 0; k0 < K; k0 += 32){
    __syncthreads();
    stage16B(&A [(m0 + srow) * K + k0 + sch], &Al[srow][sch]);   // async -> LDS
    stage16B(&Bt[(n0 + srow) * K + k0 + sch], &Bl[srow][sch]);   // async -> LDS
    stage_wait();
    __syncthreads();

    Frag af[2], bfr[4];
#pragma unroll
    for (int i = 0; i < 2; ++i)
      load_a(af[i], &Al[0][0], 32 * wm + 16 * i + n16, 40, 0, half);
#pragma unroll
    for (int j = 0; j < 4; ++j)
      load_b(bfr[j], &Bl[64 * wn + 16 * j + n16][0], half);
#pragma unroll
    for (int i = 0; i < 2; ++i)
#pragma unroll
      for (int j = 0; j < 4; ++j)
        wmma_bf16(acc[i][j], af[i], bfr[j]);
  }

#pragma unroll
  for (int i = 0; i < 2; ++i)
#pragma unroll
    for (int j = 0; j < 4; ++j)
#pragma unroll
      for (int v = 0; v < 8; ++v){
        int m = m0 + 32 * wm + 16 * i + v + 8 * half;
        int n = n0 + 64 * wn + 16 * j + n16;
        float val = (acc[i][j].f[v] + bias[n]) * alpha;
        if (mode == 0){
          ((unsigned short*)out)[(((n >> 6) * S_LEN) + m) * HDIM + (n & 63)] = f2bf(val);
        } else {
          ((float*)out)[m * D_MODEL + n] = val;
        }
      }
}

// ---------------- fused flash attention with relative positions ----------------
__global__ __launch_bounds__(128)
void attention(const unsigned short* __restrict__ Qh, const unsigned short* __restrict__ Kh,
               const unsigned short* __restrict__ Vh, const unsigned short* __restrict__ relT,
               unsigned short* __restrict__ Obuf){
  __shared__ unsigned short Kl[32][72];       // j rows x 64 d (padded)
  __shared__ unsigned short Vt[64][40];       // d rows x 32 j (transposed, padded)
  __shared__ unsigned short Pl[4][16][40];    // per-wave probability tile

  int tid  = threadIdx.x;
  int lane = tid & 31, n16 = lane & 15, half = lane >> 4;
  int w    = tid >> 5;
  int h    = blockIdx.x >> 5;
  int it   = blockIdx.x & 31;
  int i0w  = it * 64 + w * 16;

  const unsigned short* Qb = Qh + (size_t)h * S_LEN * HDIM;
  const unsigned short* Kb = Kh + (size_t)h * S_LEN * HDIM;
  const unsigned short* Vb = Vh + (size_t)h * S_LEN * HDIM;

  Frag qa[2];
  load_a(qa[0], Qb, i0w + n16, HDIM, 0,  half);
  load_a(qa[1], Qb, i0w + n16, HDIM, 32, half);

  Acc oacc[4];
#pragma unroll
  for (int t = 0; t < 4; ++t) zero_acc(oacc[t]);
  float m_run[8], l_run[8];
#pragma unroll
  for (int v = 0; v < 8; ++v){ m_run[v] = -1e30f; l_run[v] = 0.0f; }

  int krow = tid >> 2, kch = (tid & 3) * 16;  // K staging: 32 rows x 4 chunks
  int vj   = tid & 31, vg  = tid >> 5;        // V transpose staging

  const float L2E = 1.44269504088896340736f;

  for (int j0 = 0; j0 < S_LEN; j0 += 32){
    __syncthreads();
    stage16B(&Kb[(j0 + krow) * HDIM + kch], &Kl[krow][kch]);   // async -> LDS
    {
      const unsigned short* vp = Vb + (j0 + vj) * HDIM + vg * 16;
#pragma unroll
      for (int dd = 0; dd < 16; ++dd) Vt[vg * 16 + dd][vj] = vp[dd];
    }
    if (j0 + 32 < S_LEN)
      __builtin_prefetch(&Kb[(j0 + 32 + krow) * HDIM + kch], 0, 1);  // global_prefetch
    stage_wait();
    __syncthreads();

    // ---- scores: Q (16x64) x K^T (64x32), K already scaled by 1/sqrt(HD) ----
    Acc sa[2]; zero_acc(sa[0]); zero_acc(sa[1]);
#pragma unroll
    for (int nt = 0; nt < 2; ++nt)
#pragma unroll
      for (int kc = 0; kc < 2; ++kc){
        Frag kb; load_b(kb, &Kl[nt * 16 + n16][kc * 32], half);
        wmma_bf16(sa[nt], qa[kc], kb);
      }

    // ---- relative scores: Q x band-of-table, then diagonal gather ----
#pragma unroll
    for (int jt = 0; jt < 2; ++jt){
      int base2 = i0w - (j0 + 16 * jt) + S_LEN - 16;   // in [0, 4064]
      Acc qr[2]; zero_acc(qr[0]); zero_acc(qr[1]);
#pragma unroll
      for (int rt = 0; rt < 2; ++rt)
#pragma unroll
        for (int kc = 0; kc < 2; ++kc){
          Frag tb; load_b(tb, relT + (size_t)(base2 + rt * 16 + n16) * HDIM + kc * 32, half);
          wmma_bf16(qr[rt], qa[kc], tb);
        }
      // rel[r][c] = qrel_local[r][r-c+15]; same vgpr, same lane half, lane = n'&15
#pragma unroll
      for (int v = 0; v < 8; ++v){
        int r   = v + 8 * half;
        int np  = r - n16 + 15;                 // 0..30
        int src = (np & 15) | (lane & 16);
        float g0 = __shfl(qr[0].f[v], src, 32);
        float g1 = __shfl(qr[1].f[v], src, 32);
        sa[jt].f[v] += (np < 16) ? g0 : g1;
      }
    }

    // ---- online softmax ----
    float alpha_r[8];
#pragma unroll
    for (int v = 0; v < 8; ++v){
      float t = fmaxf(sa[0].f[v], sa[1].f[v]);
      t = fmaxf(t, __shfl_xor(t, 1, 32));
      t = fmaxf(t, __shfl_xor(t, 2, 32));
      t = fmaxf(t, __shfl_xor(t, 4, 32));
      t = fmaxf(t, __shfl_xor(t, 8, 32));
      float mn = fmaxf(m_run[v], t);
      alpha_r[v] = exp2f((m_run[v] - mn) * L2E);
      m_run[v]   = mn;
      sa[0].f[v] = exp2f((sa[0].f[v] - mn) * L2E);
      sa[1].f[v] = exp2f((sa[1].f[v] - mn) * L2E);
      float rs = sa[0].f[v] + sa[1].f[v];
      rs += __shfl_xor(rs, 1, 32);
      rs += __shfl_xor(rs, 2, 32);
      rs += __shfl_xor(rs, 4, 32);
      rs += __shfl_xor(rs, 8, 32);
      l_run[v] = l_run[v] * alpha_r[v] + rs;
    }
#pragma unroll
    for (int nt = 0; nt < 4; ++nt)
#pragma unroll
      for (int v = 0; v < 8; ++v) oacc[nt].f[v] *= alpha_r[v];

    // ---- P (C-layout) -> LDS -> A-fragment; then P x V ----
#pragma unroll
    for (int nt = 0; nt < 2; ++nt)
#pragma unroll
      for (int v = 0; v < 8; ++v)
        Pl[w][v + 8 * half][nt * 16 + n16] = f2bf(sa[nt].f[v]);

    Frag pa; load_a(pa, &Pl[w][0][0], n16, 40, 0, half);
#pragma unroll
    for (int nt = 0; nt < 4; ++nt){
      Frag vb2; load_b(vb2, &Vt[nt * 16 + n16][0], half);
      wmma_bf16(oacc[nt], pa, vb2);
    }
  }

  // ---- epilogue: normalize, store [s][h*64+d] bf16 ----
#pragma unroll
  for (int v = 0; v < 8; ++v) l_run[v] = 1.0f / l_run[v];
#pragma unroll
  for (int nt = 0; nt < 4; ++nt)
#pragma unroll
    for (int v = 0; v < 8; ++v){
      int m   = i0w + v + 8 * half;
      int col = h * HDIM + nt * 16 + n16;
      Obuf[(size_t)m * D_MODEL + col] = f2bf(oacc[nt].f[v] * l_run[v]);
    }
}

// ---------------- host launch ----------------

extern "C" void kernel_launch(void* const* d_in, const int* in_sizes, int n_in,
                              void* d_out, int out_size, void* d_ws, size_t ws_size,
                              hipStream_t stream) {
  (void)in_sizes; (void)n_in; (void)out_size; (void)ws_size;
  const float* x   = (const float*)d_in[0];
  const float* Wq  = (const float*)d_in[1];
  const float* bq  = (const float*)d_in[2];
  const float* Wk  = (const float*)d_in[3];
  const float* bk  = (const float*)d_in[4];
  const float* Wv  = (const float*)d_in[5];
  const float* bv  = (const float*)d_in[6];
  const float* Wo  = (const float*)d_in[7];
  const float* bo  = (const float*)d_in[8];
  const float* rel = (const float*)d_in[9];

  char* ws = (char*)d_ws;
  size_t off = 0;
  auto carve = [&](size_t bytes) -> char* {
    char* p = ws + off;
    off = (off + bytes + 255) & ~(size_t)255;
    return p;
  };
  const size_t SD = (size_t)S_LEN * D_MODEL;
  unsigned short* x_bf  = (unsigned short*)carve(SD * 2);      // reused as Obuf later
  unsigned short* Wt_q  = (unsigned short*)carve((size_t)D_MODEL * D_MODEL * 2);
  unsigned short* Wt_k  = (unsigned short*)carve((size_t)D_MODEL * D_MODEL * 2);
  unsigned short* Wt_v  = (unsigned short*)carve((size_t)D_MODEL * D_MODEL * 2);
  unsigned short* Wt_o  = (unsigned short*)carve((size_t)D_MODEL * D_MODEL * 2);
  unsigned short* Qh    = (unsigned short*)carve(SD * 2);
  unsigned short* Kh    = (unsigned short*)carve(SD * 2);
  unsigned short* Vh    = (unsigned short*)carve(SD * 2);
  unsigned short* relbf = (unsigned short*)carve((size_t)4096 * HDIM * 2);
  unsigned short* Obuf  = x_bf;  // alias: x_bf dead after projections

  cvt_bf16<<<(int)(SD / 256), 256, 0, stream>>>(x, x_bf, (int)SD);
  cvt_transpose<<<(D_MODEL * D_MODEL) / 256, 256, 0, stream>>>(Wq, Wt_q);
  cvt_transpose<<<(D_MODEL * D_MODEL) / 256, 256, 0, stream>>>(Wk, Wt_k);
  cvt_transpose<<<(D_MODEL * D_MODEL) / 256, 256, 0, stream>>>(Wv, Wt_v);
  cvt_transpose<<<(D_MODEL * D_MODEL) / 256, 256, 0, stream>>>(Wo, Wt_o);
  cvt_rel<<<(4096 * HDIM) / 256, 256, 0, stream>>>(rel, relbf);

  gemm2048<<<128, 256, 0, stream>>>(x_bf, Wt_q, bq, Qh, 1.0f,   0);
  gemm2048<<<128, 256, 0, stream>>>(x_bf, Wt_k, bk, Kh, 0.125f, 0);  // fold 1/sqrt(64)
  gemm2048<<<128, 256, 0, stream>>>(x_bf, Wt_v, bv, Vh, 1.0f,   0);

  attention<<<NHEADS * (S_LEN / 64), 128, 0, stream>>>(Qh, Kh, Vh, relbf, Obuf);

  gemm2048<<<128, 256, 0, stream>>>(Obuf, Wt_o, bo, (float*)d_out, 1.0f, 1);
}
